// RGCNEncoder_4861902979425
// MI455X (gfx1250) — matrix operand order repaired and checked by reading, compiled, not run on hardware
//
#include <hip/hip_runtime.h>

// RGCN encoder for MI455X (gfx1250, wave32).
// Roofline: edge scatter moves ~13.6 GB (bound, ~0.6ms @ 23.3 TB/s);
// the 11 N x 128 x 128 GEMMs (~36 GFLOP) run on v_wmma_f32_16x16x32_bf16
// with weights pre-packed to fragment order and staged in LDS (32 KB/WGP).

#define D       128
#define NBASES  4
#define NMAT    11            // 8 relations + self-loop + 2 NGNN layers
#define MAT_ELEMS (D * D)     // 16384 packed bf16 elements per matrix

typedef __attribute__((ext_vector_type(8)))  float  v8f;
typedef __attribute__((ext_vector_type(16))) __bf16 v16bf;

// gemm flags
#define F_DEG  1   // normalize A rows by 1/max(deg,1)
#define F_ACC  2   // accumulate into existing Out (reads Cin)
#define F_RELU 4   // relu epilogue
#define F_BIAS 8   // add per-column bias

__global__ void zero_kernel(float* __restrict__ p, long n) {
  long i = (long)blockIdx.x * blockDim.x + threadIdx.x;
  long stride = (long)gridDim.x * blockDim.x;
  for (; i < n; i += stride) p[i] = 0.0f;
}

// Pack all 11 weight matrices into WMMA B-fragment order, bf16:
// layout [mat][kb(4)][ct(8)][lane(32)][e(16)]; per-lane 32B contiguous.
// B-fragment mapping (contiguous k order, same as A): col = ct*16 + (lane&15),
// k = kb*32 + (lane>>4)*16 + e.
__global__ void pack_weights_kernel(const float* __restrict__ w_comp,
                                    const float* __restrict__ bases,
                                    const float* __restrict__ loop_w,
                                    const float* __restrict__ ngnn_w,
                                    __bf16* __restrict__ Wpack) {
  int t = blockIdx.x * blockDim.x + threadIdx.x;
  if (t >= NMAT * MAT_ELEMS) return;
  int m    = t >> 14;
  int rem  = t & (MAT_ELEMS - 1);
  int e    = rem & 15;
  int lane = (rem >> 4) & 31;
  int ct   = (rem >> 9) & 7;
  int kb   = rem >> 12;
  int khalf = lane >> 4;
  int n = ct * 16 + (lane & 15);
  int k = kb * 32 + khalf * 16 + e;
  float v;
  if (m < 8) {                       // basis-decomposed relation weights
    v = 0.0f;
#pragma unroll
    for (int b = 0; b < NBASES; ++b)
      v += w_comp[m * NBASES + b] * bases[(b * D + k) * D + n];
  } else if (m == 8) {               // self-loop weight
    v = loop_w[k * D + n];
  } else {                           // NGNN layers
    v = ngnn_w[(m - 9) * (D * D) + k * D + n];
  }
  Wpack[t] = (__bf16)v;
}

// One wave per edge (grid-stride): lanes cover the 128-float row as float4,
// scatter via global_atomic_add_f32; lane 0 counts in-degree.
__global__ void scatter_kernel(const float* __restrict__ x,
                               const int* __restrict__ src,
                               const int* __restrict__ dst,
                               float* __restrict__ msg,
                               float* __restrict__ deg,
                               int nEdges) {
  int lane   = threadIdx.x & 31;
  int wave   = blockIdx.x * (blockDim.x >> 5) + (threadIdx.x >> 5);
  int nWaves = gridDim.x * (blockDim.x >> 5);
  for (int e = wave; e < nEdges; e += nWaves) {
    int s = src[e];
    int d = dst[e];
    const float4 v = reinterpret_cast<const float4*>(x + (long)s * D)[lane];
    float* o = msg + (long)d * D + lane * 4;
    atomicAdd(o + 0, v.x);
    atomicAdd(o + 1, v.y);
    atomicAdd(o + 2, v.z);
    atomicAdd(o + 3, v.w);
    if (lane == 0) atomicAdd(deg + d, 1.0f);
  }
}

// Out[n,128] (+)= (A[n,128] * 1/deg?) @ W (packed bf16), fused bias/relu.
// Block = 256 threads = 8 waves; each wave owns a 16x128 output strip.
// Weight matrix (32 KB bf16) is staged in LDS once per block.
__global__ __launch_bounds__(256)
void gemm_bf16_kernel(const float* __restrict__ A,
                      const float* __restrict__ deg,
                      const __bf16* __restrict__ Wpack,
                      const float* __restrict__ Cin,
                      const float* __restrict__ bias,
                      float* __restrict__ Out,
                      int nRows, int flags) {
  __shared__ __align__(32) __bf16 lds[MAT_ELEMS];
  {
    const uint4* g = reinterpret_cast<const uint4*>(Wpack);
    uint4*       l = reinterpret_cast<uint4*>(lds);
#pragma unroll
    for (int i = 0; i < (MAT_ELEMS * 2 / 16) / 256; ++i)   // 2048 uint4 total
      l[threadIdx.x + i * 256] = g[threadIdx.x + i * 256];
  }
  __syncthreads();

  int lane  = threadIdx.x & 31;
  int wid   = threadIdx.x >> 5;
  int khalf = lane >> 4;
  long rowbase = (long)blockIdx.x * 128 + wid * 16;
  long myrow   = rowbase + (lane & 15);
  bool rv = myrow < (long)nRows;
  long arow = (rv ? myrow : (long)(nRows - 1)) * D;

  float scale = 1.0f;
  if (flags & F_DEG) {
    float dg = rv ? deg[myrow] : 1.0f;
    scale = 1.0f / fmaxf(dg, 1.0f);
  }
  if (!rv) scale = 0.0f;   // zero contribution of padded rows (EXEC stays all-1)

  v8f zero = {};
  v8f acc[8];
#pragma unroll
  for (int ct = 0; ct < 8; ++ct) acc[ct] = zero;

  const float4* ar = reinterpret_cast<const float4*>(A + arow);
#pragma unroll
  for (int kb = 0; kb < 4; ++kb) {
    // A-fragment (contiguous k order, same as packed B): k = khalf*16 + e
    int b0 = kb * 8 + khalf * 4;
    float4 f0 = ar[b0], f1 = ar[b0 + 1], f2 = ar[b0 + 2], f3 = ar[b0 + 3];
    v16bf a;
    a[0]  = (__bf16)(f0.x * scale); a[1]  = (__bf16)(f0.y * scale);
    a[2]  = (__bf16)(f0.z * scale); a[3]  = (__bf16)(f0.w * scale);
    a[4]  = (__bf16)(f1.x * scale); a[5]  = (__bf16)(f1.y * scale);
    a[6]  = (__bf16)(f1.z * scale); a[7]  = (__bf16)(f1.w * scale);
    a[8]  = (__bf16)(f2.x * scale); a[9]  = (__bf16)(f2.y * scale);
    a[10] = (__bf16)(f2.z * scale); a[11] = (__bf16)(f2.w * scale);
    a[12] = (__bf16)(f3.x * scale); a[13] = (__bf16)(f3.y * scale);
    a[14] = (__bf16)(f3.z * scale); a[15] = (__bf16)(f3.w * scale);
#pragma unroll
    for (int ct = 0; ct < 8; ++ct) {
      const v16bf b =
          *reinterpret_cast<const v16bf*>(&lds[((kb * 8 + ct) * 32 + lane) * 16]);
      acc[ct] = __builtin_amdgcn_wmma_f32_16x16x32_bf16(
          false, a, false, b, (short)0, acc[ct], false, false);
    }
  }

  // Epilogue: C/D layout -> row = rowbase + j + khalf*8, col = ct*16 + (lane&15)
#pragma unroll
  for (int ct = 0; ct < 8; ++ct) {
    int col = ct * 16 + (lane & 15);
#pragma unroll
    for (int j = 0; j < 8; ++j) {
      long row = rowbase + j + khalf * 8;
      if (row < (long)nRows) {
        float v = acc[ct][j];
        long idx = row * D + col;
        if (flags & F_ACC)  v += Cin[idx];
        if (flags & F_BIAS) v += bias[col];
        if (flags & F_RELU) v = fmaxf(v, 0.0f);
        Out[idx] = v;
      }
    }
  }
}

extern "C" void kernel_launch(void* const* d_in, const int* in_sizes, int n_in,
                              void* d_out, int out_size, void* d_ws, size_t ws_size,
                              hipStream_t stream) {
  const float* x     = (const float*)d_in[0];
  const int*   esrc  = (const int*)  d_in[1];
  const int*   edst  = (const int*)  d_in[2];
  const float* wcomp = (const float*)d_in[3];
  const float* bases = (const float*)d_in[4];
  const float* loopw = (const float*)d_in[5];
  const float* hbias = (const float*)d_in[6];
  const float* ngnnw = (const float*)d_in[7];
  float* out = (float*)d_out;

  const int N = in_sizes[0] / D;          // 100000
  const int R = in_sizes[3] / NBASES;     // 8
  const int E = in_sizes[1] / R;          // 800000

  // Workspace layout: [Wpack 352KB (pad to 512KB)] [h N*D f32] [msg N*D f32] [deg N f32]
  char*   ws    = (char*)d_ws;
  __bf16* wpack = (__bf16*)ws;
  float*  h     = (float*)(ws + (512 << 10));
  float*  msg   = h + (size_t)N * D;
  float*  deg   = msg + (size_t)N * D;    // contiguous after msg

  dim3 blk(256);
  const int gemmGrid = (N + 127) / 128;

  // 1. pack all weights to bf16 fragment order
  pack_weights_kernel<<<dim3((NMAT * MAT_ELEMS + 255) / 256), blk, 0, stream>>>(
      wcomp, bases, loopw, ngnnw, wpack);

  // 2. h = x @ loop_weight
  gemm_bf16_kernel<<<dim3(gemmGrid), blk, 0, stream>>>(
      x, nullptr, wpack + 8 * MAT_ELEMS, nullptr, nullptr, h, N, 0);

  // 3. per-relation: scatter mean-aggregate, then h += agg @ W_r
  //    (bias + relu fused into the last relation's epilogue)
  for (int r = 0; r < R; ++r) {
    zero_kernel<<<dim3(1024), blk, 0, stream>>>(msg, (long)N * (D + 1));
    scatter_kernel<<<dim3(2048), blk, 0, stream>>>(
        x, esrc + (size_t)r * E, edst + (size_t)r * E, msg, deg, E);
    int fl = F_DEG | F_ACC | ((r == R - 1) ? (F_BIAS | F_RELU) : 0);
    gemm_bf16_kernel<<<dim3(gemmGrid), blk, 0, stream>>>(
        msg, deg, wpack + r * MAT_ELEMS, h, hbias, h, N, fl);
  }

  // 4. NGNN MLP: ping-pong h -> msg -> out
  gemm_bf16_kernel<<<dim3(gemmGrid), blk, 0, stream>>>(
      h, nullptr, wpack + 9 * MAT_ELEMS, nullptr, nullptr, msg, N, F_RELU);
  gemm_bf16_kernel<<<dim3(gemmGrid), blk, 0, stream>>>(
      msg, nullptr, wpack + 10 * MAT_ELEMS, nullptr, nullptr, out, N, F_RELU);
}
